// FusedThorExpert_69380901700281
// MI455X (gfx1250) — matrix operand order repaired and checked by reading, not compiled, not run on hardware
//
#include <hip/hip_runtime.h>
#include <hip/hip_bf16.h>

typedef __attribute__((ext_vector_type(16))) _Float16 v16h;
typedef __attribute__((ext_vector_type(8)))  _Float16 v8h;
typedef __attribute__((ext_vector_type(4)))  _Float16 v4h;
typedef __attribute__((ext_vector_type(8)))  float    v8f;
typedef __attribute__((ext_vector_type(4)))  float    v4f;

namespace {
constexpr int kE = 8, kCAP = 2048, kH = 1024, kI = 4096;
constexpr int BM = 128, BN = 256, BK = 32;
constexpr int PITCH = 40;  // halfs per LDS row: 32 data + 8 pad (80B rows: b128-aligned, conflict-free)
}

static __device__ __forceinline__ v16h cat8(v8h a, v8h b) {
  v16h r;
#pragma unroll
  for (int i = 0; i < 8; ++i) { r[i] = a[i]; r[i + 8] = b[i]; }
  return r;
}

// fp32 -> f16 (hi, lo) split of 4 lanes. hi uses RTZ pack (1 VALU / 2 elems);
// lo is the exact residual vs the rounded hi, so total err ~2^-21 regardless.
static __device__ __forceinline__ void split4(const v4f x, v4h& hi, v4h& lo) {
#if __has_builtin(__builtin_amdgcn_cvt_pkrtz)
  const auto h01 = __builtin_amdgcn_cvt_pkrtz(x[0], x[1]);  // __fp16 vec2
  const auto h23 = __builtin_amdgcn_cvt_pkrtz(x[2], x[3]);
  const float r0 = x[0] - (float)h01[0];
  const float r1 = x[1] - (float)h01[1];
  const float r2 = x[2] - (float)h23[0];
  const float r3 = x[3] - (float)h23[1];
  const auto l01 = __builtin_amdgcn_cvt_pkrtz(r0, r1);
  const auto l23 = __builtin_amdgcn_cvt_pkrtz(r2, r3);
  hi[0] = (_Float16)h01[0]; hi[1] = (_Float16)h01[1];
  hi[2] = (_Float16)h23[0]; hi[3] = (_Float16)h23[1];
  lo[0] = (_Float16)l01[0]; lo[1] = (_Float16)l01[1];
  lo[2] = (_Float16)l23[0]; lo[3] = (_Float16)l23[1];
#else
#pragma unroll
  for (int c = 0; c < 4; ++c) {
    const _Float16 h = (_Float16)x[c];
    hi[c] = h;
    lo[c] = (_Float16)(x[c] - (float)h);
  }
#endif
}

// Grouped GEMM, fp32 emulated via f16 hi/lo split (3 WMMA products):
// out[e,m,n] = sum_k X[e,m,k] * W[e,n,k] + b[e,n]
__global__ __launch_bounds__(256, 1)
void moe_grouped_gemm_f16x3(const float* __restrict__ X,
                            const float* __restrict__ W,
                            const float* __restrict__ B1,
                            float* __restrict__ Out)
{
  __shared__ _Float16 sAhi[BM * PITCH];
  __shared__ _Float16 sAlo[BM * PITCH];
  __shared__ _Float16 sBhi[BN * PITCH];
  __shared__ _Float16 sBlo[BN * PITCH];

  constexpr int ntN = kI / BN;       // 16
  constexpr int ntM = kCAP / BM;     // 16
  constexpr int perExp = ntM * ntN;  // 256

  const int wg = blockIdx.x;
  const int e  = wg / perExp;
  const int tt = wg % perExp;
  const int mt = tt / ntN;
  const int nt = tt % ntN;

  const float* __restrict__ Xe = X + (size_t)(e * kCAP + mt * BM) * kH;
  const float* __restrict__ We = W + ((size_t)e * kI + (size_t)nt * BN) * kH;

  const int tid  = threadIdx.x;
  const int wave = tid >> 5;        // 8 waves (wave32)
  const int lane = tid & 31;
  const int wm   = wave >> 1;       // 0..3 -> row offset wm*32
  const int wn   = wave & 1;        // 0..1 -> col offset wn*128
  const int lr   = lane & 15;       // row-within-fragment
  const int ls   = lane >> 4;       // lane-half select (K group)

  v8f acc[2][8];
#pragma unroll
  for (int i = 0; i < 2; ++i)
#pragma unroll
    for (int j = 0; j < 8; ++j) { v8f z{}; acc[i][j] = z; }

  for (int kb = 0; kb < kH; kb += BK) {
    __syncthreads();
    // ---- Stage X tile (128x32): 256 threads x 4 float4 ----
#pragma unroll
    for (int j = 0; j < 4; ++j) {
      const int id  = tid + 256 * j;
      const int row = id >> 3;              // 0..127
      const int c4  = (id & 7) << 2;        // 0,4,...,28
      const v4f x = *(const v4f*)(Xe + (size_t)row * kH + kb + c4);
      v4h h, l;
      split4(x, h, l);
      *(v4h*)&sAhi[row * PITCH + c4] = h;
      *(v4h*)&sAlo[row * PITCH + c4] = l;
    }
    // ---- Stage W tile (256x32): 256 threads x 8 float4 ----
#pragma unroll
    for (int j = 0; j < 8; ++j) {
      const int id  = tid + 256 * j;
      const int row = id >> 3;              // 0..255
      const int c4  = (id & 7) << 2;
      const v4f x = *(const v4f*)(We + (size_t)row * kH + kb + c4);
      v4h h, l;
      split4(x, h, l);
      *(v4h*)&sBhi[row * PITCH + c4] = h;
      *(v4h*)&sBlo[row * PITCH + c4] = l;
    }
    // Speculative prefetch of next K-tile into GL2 (global_prefetch_b8).
    if (kb + BK < kH) {
      const int row = tid >> 3;
      const int c4  = (tid & 7) << 2;
      __builtin_prefetch(Xe + (size_t)row * kH + kb + BK + c4, 0, 1);
      __builtin_prefetch(We + (size_t)(row + 128) * kH + kb + BK + c4, 0, 1);
    }
    __syncthreads();

    // ---- A fragments (held): lane = M row; elems 0..7 = K[8*ls..], 8..15 = K[16+8*ls..] ----
    v16h aHi[2], aLo[2];
#pragma unroll
    for (int mf = 0; mf < 2; ++mf) {
      const int m = wm * 32 + mf * 16 + lr;
      const _Float16* p = &sAhi[m * PITCH + ls * 8];
      aHi[mf] = cat8(*(const v8h*)p, *(const v8h*)(p + 16));
      const _Float16* q = &sAlo[m * PITCH + ls * 8];
      aLo[mf] = cat8(*(const v8h*)q, *(const v8h*)(q + 16));
    }

    // ---- Stream B fragments; 6 WMMAs each (hi*hi + hi*lo + lo*hi) ----
#pragma unroll
    for (int nf = 0; nf < 8; ++nf) {
      const int n = wn * 128 + nf * 16 + lr;
      const _Float16* p = &sBhi[n * PITCH + ls * 8];
      const v16h bHi = cat8(*(const v8h*)p, *(const v8h*)(p + 16));
      const _Float16* q = &sBlo[n * PITCH + ls * 8];
      const v16h bLo = cat8(*(const v8h*)q, *(const v8h*)(q + 16));
#pragma unroll
      for (int mf = 0; mf < 2; ++mf) {
        v8f c = acc[mf][nf];
        c = __builtin_amdgcn_wmma_f32_16x16x32_f16(false, aHi[mf], false, bHi,
                                                   (short)0, c, false, false);
        c = __builtin_amdgcn_wmma_f32_16x16x32_f16(false, aHi[mf], false, bLo,
                                                   (short)0, c, false, false);
        c = __builtin_amdgcn_wmma_f32_16x16x32_f16(false, aLo[mf], false, bHi,
                                                   (short)0, c, false, false);
        acc[mf][nf] = c;
      }
    }
  }

  // ---- Epilogue: bias add + store (C/D: lane%16 = N col, VGPR r = M row, +8 for lanes 16..31) ----
#pragma unroll
  for (int mf = 0; mf < 2; ++mf)
#pragma unroll
    for (int nf = 0; nf < 8; ++nf) {
      const int n = nt * BN + wn * 128 + nf * 16 + lr;
      const float bias = B1[e * kI + n];
      const int m0 = mt * BM + wm * 32 + mf * 16 + 8 * ls;
#pragma unroll
      for (int r = 0; r < 8; ++r) {
        const size_t row = (size_t)(e * kCAP + m0 + r);
        Out[row * (size_t)kI + n] = acc[mf][nf][r] + bias;
      }
    }
}

extern "C" void kernel_launch(void* const* d_in, const int* in_sizes, int n_in,
                              void* d_out, int out_size, void* d_ws, size_t ws_size,
                              hipStream_t stream) {
  (void)in_sizes; (void)n_in; (void)d_ws; (void)ws_size; (void)out_size;
  const float* X  = (const float*)d_in[0];   // inter_state [E*CAP, H]
  const float* W  = (const float*)d_in[1];   // W1 [E, I, H]
  const float* B1 = (const float*)d_in[2];   // b1 [E, I]
  float* Out = (float*)d_out;                // [E*CAP, I]

  constexpr int blocks = kE * (kCAP / BM) * (kI / BN);  // 8*16*16 = 2048
  moe_grouped_gemm_f16x3<<<dim3(blocks), dim3(256), 0, stream>>>(X, W, B1, Out);
}
